// Classifier_33646773797257
// MI455X (gfx1250) — matrix operand order repaired and checked by reading, not compiled, not run on hardware
//
#include <hip/hip_runtime.h>
#include <hip/hip_bf16.h>

typedef __attribute__((ext_vector_type(16))) __bf16 bf16x16;
typedef __attribute__((ext_vector_type(8)))  float  v8f;

#define HD 256   // H*D
#define HH 4     // heads
#define DD 64    // per-head dim
#define GG 64    // graphs
#define CC 10    // classes
#define TN 8     // output tiles along N per wave (16x128 per wave)
#define SCAN_B 256

// ---------- pack W[K,Ncols] f32 -> bf16 WMMA B-fragments, k-block-major ----------
// Bp index p = ((kb*TilesN + n16)*32 + lane)*16 + e
// maps to W[k, col], k = kb*32 + (lane>=16 ? 8 : 0) + (e&7) + (e>>3)*16, col = n16*16 + (lane&15)
__global__ void pack_b_kernel(const float* __restrict__ W, __bf16* __restrict__ Bp,
                              int K, int Ncols) {
    int p = blockIdx.x * blockDim.x + threadIdx.x;
    if (p >= K * Ncols) return;
    int TNt  = Ncols >> 4;
    int e    = p & 15;
    int lane = (p >> 4) & 31;
    int t    = p >> 9;           // kb*TNt + n16
    int n16  = t % TNt;
    int kb   = t / TNt;
    int k    = kb * 32 + ((lane >= 16) ? 8 : 0) + (e & 7) + ((e >> 3) << 4);
    int col  = n16 * 16 + (lane & 15);
    Bp[p] = (__bf16)W[(size_t)k * Ncols + col];
}

// ---------- bf16 WMMA GEMM: C[M,Ncols] = A[M,K] @ W, one wave per 16x(16*TN) tile ----------
__global__ void __launch_bounds__(256, 2)
gemm_bf16_wmma(const float* __restrict__ A,
               const __bf16* __restrict__ Bp,
               float* __restrict__ C,
               int M, int K, int Ncols) {
    int wave = (blockIdx.x * blockDim.x + threadIdx.x) >> 5;
    int lane = threadIdx.x & 31;
    int TG    = Ncols / (16 * TN);         // tile-groups along N
    int TNt   = Ncols >> 4;                // total 16-col tiles
    int tileM = wave / TG;
    int tg    = wave - tileM * TG;
    if (tileM * 16 >= M) return;           // wave-uniform

    const bool hi   = lane >= 16;
    const int  l15  = lane & 15;
    const int  rowA = tileM * 16 + l15;
    const int  KB   = K >> 5;

    v8f acc[TN];
#pragma unroll
    for (int j = 0; j < TN; ++j) acc[j] = (v8f){};

    const float* arow = A + (size_t)rowA * K + (hi ? 8 : 0);
    for (int kb = 0; kb < KB; ++kb) {
        // all TN B fragments for this k-block sit at base + j*512 elements (constant offsets)
        const __bf16* bbase = Bp + ((size_t)(kb * TNt + tg * TN) * 32 + lane) * 16;

        // A tile 16x32 bf16: element e -> K = kb*32 + (hi?8:0) + (e&7) + (e>>3)*16
        bf16x16 a;
#pragma unroll
        for (int e = 0; e < 16; ++e) {
            int kk = (e & 7) + ((e >> 3) << 4);
            a[e] = (__bf16)arow[kb * 32 + kk];
        }

        {
            bf16x16 bf[4];
#pragma unroll
            for (int j = 0; j < 4; ++j) bf[j] = *(const bf16x16*)(bbase + j * 512);
#pragma unroll
            for (int j = 0; j < 4; ++j) {
                acc[j] = __builtin_amdgcn_wmma_f32_16x16x32_bf16(
                    false, a, false, bf[j], (short)0, acc[j], false, false);
            }
        }
        {
            bf16x16 bf[4];
#pragma unroll
            for (int j = 0; j < 4; ++j) bf[j] = *(const bf16x16*)(bbase + (4 + j) * 512);
#pragma unroll
            for (int j = 0; j < 4; ++j) {
                acc[4 + j] = __builtin_amdgcn_wmma_f32_16x16x32_bf16(
                    false, a, false, bf[j], (short)0, acc[4 + j], false, false);
            }
        }
    }
    // C/D layout: VGPR r -> M = r (+8 for hi lanes), N = lane&15
    int mbase = tileM * 16 + (hi ? 8 : 0);
#pragma unroll
    for (int j = 0; j < TN; ++j) {
        int n = (tg * TN + j) * 16 + l15;
#pragma unroll
        for (int r = 0; r < 8; ++r) {
            C[(size_t)(mbase + r) * Ncols + n] = acc[j][r];
        }
    }
}

// ---------- per-node attention coefficients: el/er [N,H] ----------
__global__ void attn_coef_kernel(const float* __restrict__ feat,
                                 const float* __restrict__ al,
                                 const float* __restrict__ ar,
                                 float* __restrict__ el,
                                 float* __restrict__ er) {
    __shared__ float sl[HD];
    __shared__ float sr[HD];
    int n = blockIdx.x;
    int t = threadIdx.x;          // 256 threads
    int h = t >> 6, d = t & 63;
    float f = feat[(size_t)n * HD + t];
    sl[t] = f * al[h * DD + d];
    sr[t] = f * ar[h * DD + d];
    __syncthreads();
    for (int s = 32; s >= 1; s >>= 1) {
        if (d < s) { sl[t] += sl[t + s]; sr[t] += sr[t + s]; }
        __syncthreads();
    }
    if (d == 0) { el[n * HH + h] = sl[t]; er[n * HH + h] = sr[t]; }
}

// ================= CSR-by-dst build (once per launch; dst shared by all layers) =================
__global__ void count_kernel(const int* __restrict__ dst, int* __restrict__ counts, int E) {
    int i = blockIdx.x * blockDim.x + threadIdx.x;
    if (i < E) atomicAdd(&counts[dst[i]], 1);
}

__global__ void scan1_kernel(const int* __restrict__ counts, int* __restrict__ rowptr,
                             int* __restrict__ bsum, int N) {
    __shared__ int sh[SCAN_B];
    int t = threadIdx.x, i = blockIdx.x * SCAN_B + t;
    int v = (i < N) ? counts[i] : 0;
    sh[t] = v;
    __syncthreads();
    for (int o = 1; o < SCAN_B; o <<= 1) {
        int add = (t >= o) ? sh[t - o] : 0;
        __syncthreads();
        sh[t] += add;
        __syncthreads();
    }
    if (i < N) rowptr[i] = sh[t] - v;          // exclusive scan
    if (t == SCAN_B - 1) bsum[blockIdx.x] = sh[t];
}

__global__ void scan2_kernel(int* __restrict__ bsum, int nb) {
    if (threadIdx.x == 0) {
        int acc = 0;
        for (int i = 0; i < nb; ++i) { int v = bsum[i]; bsum[i] = acc; acc += v; }
    }
}

__global__ void scan3_kernel(int* __restrict__ rowptr, const int* __restrict__ bsum, int N) {
    int i = blockIdx.x * SCAN_B + threadIdx.x;
    if (i < N) rowptr[i] += bsum[blockIdx.x];
}

__global__ void fill_kernel(const int* __restrict__ src, const int* __restrict__ dst,
                            const int* __restrict__ rowptr, int* __restrict__ fill,
                            int* __restrict__ csr_src, int E) {
    int i = blockIdx.x * blockDim.x + threadIdx.x;
    if (i >= E) return;
    int d = dst[i];
    int pos = rowptr[d] + atomicAdd(&fill[d], 1);
    csr_src[pos] = src[i];
}

// ================= fused edge-softmax + aggregate + residual + ELU =================
// one wave per (node, head); no atomics, output written exactly once
__global__ void gat_attn_aggregate(const int* __restrict__ rowptr, const int* __restrict__ counts,
                                   const int* __restrict__ csr_src,
                                   const float* __restrict__ feat,
                                   const float* __restrict__ el, const float* __restrict__ er,
                                   const float* __restrict__ resid, float* __restrict__ outb,
                                   int N, int residual, int activate) {
    int wid  = (blockIdx.x * blockDim.x + threadIdx.x) >> 5;
    int lane = threadIdx.x & 31;
    if (wid >= N * HH) return;
    int n = wid >> 2, h = wid & 3;
    int start = rowptr[n], deg = counts[n];
    float ern = er[n * HH + h];

    // pass 1: segment max over incoming edges
    float m = -3.402823466e38f;
    for (int i = lane; i < deg; i += 32) {
        int s = csr_src[start + i];
        float v = el[s * HH + h] + ern;
        v = v > 0.f ? v : 0.2f * v;
        m = fmaxf(m, v);
    }
#pragma unroll
    for (int o = 16; o >= 1; o >>= 1) m = fmaxf(m, __shfl_xor(m, o, 32));

    // pass 2: denominator
    float dsum = 0.f;
    for (int i = lane; i < deg; i += 32) {
        int s = csr_src[start + i];
        float v = el[s * HH + h] + ern;
        v = v > 0.f ? v : 0.2f * v;
        dsum += __expf(v - m);
    }
#pragma unroll
    for (int o = 16; o >= 1; o >>= 1) dsum += __shfl_xor(dsum, o, 32);
    float inv = 1.f / fmaxf(dsum, 1e-9f);

    // pass 3: weighted aggregation, lanes cover the 64 dims of this head
    float a0 = 0.f, a1 = 0.f;
    for (int i = 0; i < deg; ++i) {
        int s = csr_src[start + i];            // uniform -> broadcast
        float v = el[s * HH + h] + ern;
        v = v > 0.f ? v : 0.2f * v;
        float alpha = __expf(v - m) * inv;
        const float* fr = feat + (size_t)s * HD + h * DD;
        a0 += alpha * fr[lane];
        a1 += alpha * fr[lane + 32];
    }
    size_t ob = (size_t)n * HD + h * DD;
    if (residual) { a0 += resid[ob + lane]; a1 += resid[ob + lane + 32]; }
    if (activate) {
        a0 = a0 > 0.f ? a0 : __expf(a0) - 1.f;
        a1 = a1 > 0.f ? a1 : __expf(a1) - 1.f;
    }
    outb[ob + lane]      = a0;
    outb[ob + lane + 32] = a1;
}

// ---------- head-mean + graph segment-sum ----------
__global__ void readout_kernel(const float* __restrict__ h2, const int* __restrict__ gid,
                               float* __restrict__ hg, float* __restrict__ cnt) {
    int n = blockIdx.x;
    int d = threadIdx.x;          // 64 threads
    float s = 0.f;
#pragma unroll
    for (int h = 0; h < HH; ++h) s += h2[(size_t)n * HD + h * DD + d];
    s *= 0.25f;
    int g = gid[n];
    atomicAdd(&hg[g * DD + d], s);
    if (d == 0) atomicAdd(&cnt[g], 1.0f);
}

// ---------- classifier: out[g,c] = (hg[g,:]/cnt) @ Wc + bc ----------
__global__ void classify_kernel(const float* __restrict__ hg, const float* __restrict__ cnt,
                                const float* __restrict__ Wc, const float* __restrict__ bc,
                                float* __restrict__ out) {
    int i = blockIdx.x * blockDim.x + threadIdx.x;
    if (i >= GG * CC) return;
    int g = i / CC, c = i - g * CC;
    float inv = 1.f / fmaxf(cnt[g], 1.f);
    float acc = bc[c];
    for (int d = 0; d < DD; ++d) acc += (hg[g * DD + d] * inv) * Wc[d * CC + c];
    out[i] = acc;
}

extern "C" void kernel_launch(void* const* d_in, const int* in_sizes, int n_in,
                              void* d_out, int out_size, void* d_ws, size_t ws_size,
                              hipStream_t stream) {
    const float* x   = (const float*)d_in[0];
    const int*   src = (const int*)d_in[1];
    const int*   dst = (const int*)d_in[2];
    const int*   gid = (const int*)d_in[3];
    const float* W0  = (const float*)d_in[4];
    const float* al0 = (const float*)d_in[5];
    const float* ar0 = (const float*)d_in[6];
    const float* W1  = (const float*)d_in[7];
    const float* al1 = (const float*)d_in[8];
    const float* ar1 = (const float*)d_in[9];
    const float* W2  = (const float*)d_in[10];
    const float* al2 = (const float*)d_in[11];
    const float* ar2 = (const float*)d_in[12];
    const float* Wc  = (const float*)d_in[13];
    const float* bc  = (const float*)d_in[14];

    const int N = in_sizes[3];      // graph_id has N elements
    const int E = in_sizes[1];      // src has E elements
    const int IN = in_sizes[0] / N; // 128
    const int NB = (N + SCAN_B - 1) / SCAN_B;

    char* ws = (char*)d_ws;
    size_t off = 0;
    auto carve = [&](size_t bytes) -> char* {
        char* p = ws + off;
        off += (bytes + 255) & ~(size_t)255;
        return p;
    };
    float*  bufA    = (float*)carve((size_t)N * HD * 4);
    float*  bufB    = (float*)carve((size_t)N * HD * 4);
    float*  bufC    = (float*)carve((size_t)N * HD * 4);
    float*  el      = (float*)carve((size_t)N * HH * 4);
    float*  er      = (float*)carve((size_t)N * HH * 4);
    __bf16* Wp      = (__bf16*)carve((size_t)HD * HD * 2);   // packed weights (max 256x256)
    int*    counts  = (int*)carve((size_t)N * 4);
    int*    rowptr  = (int*)carve((size_t)N * 4);
    int*    fillc   = (int*)carve((size_t)N * 4);
    int*    bsum    = (int*)carve((size_t)NB * 4);
    int*    csr_src = (int*)carve((size_t)E * 4);
    float*  hg      = (float*)carve(GG * DD * 4);
    float*  cnt     = (float*)carve(GG * 4);

    // ---- build CSR by dst (once; shared by all 3 layers) ----
    hipMemsetAsync(counts, 0, (size_t)N * 4, stream);
    hipMemsetAsync(fillc,  0, (size_t)N * 4, stream);
    count_kernel<<<(E + 255) / 256, 256, 0, stream>>>(dst, counts, E);
    scan1_kernel<<<NB, SCAN_B, 0, stream>>>(counts, rowptr, bsum, N);
    scan2_kernel<<<1, 32, 0, stream>>>(bsum, NB);
    scan3_kernel<<<NB, SCAN_B, 0, stream>>>(rowptr, bsum, N);
    fill_kernel<<<(E + 255) / 256, 256, 0, stream>>>(src, dst, rowptr, fillc, csr_src, E);

    auto run_layer = [&](const float* hin, int K, const float* W,
                         const float* al, const float* ar,
                         float* feat, float* aggout, const float* resid, int activate) {
        pack_b_kernel<<<(K * HD + 255) / 256, 256, 0, stream>>>(W, Wp, K, HD);
        int waves = (N / 16) * (HD / (16 * TN));   // N % 16 == 0
        gemm_bf16_wmma<<<(waves * 32) / 256, 256, 0, stream>>>(hin, Wp, feat, N, K, HD);
        attn_coef_kernel<<<N, 256, 0, stream>>>(feat, al, ar, el, er);
        int awaves = N * HH;                        // one wave per (node, head)
        gat_attn_aggregate<<<(awaves * 32 + 255) / 256, 256, 0, stream>>>(
            rowptr, counts, csr_src, feat, el, er, resid, aggout, N,
            resid != nullptr, activate);
    };

    // layer 0: x[N,128] -> bufB;  layer 1: bufB -> bufC (resid bufB);  layer 2: bufC -> bufB (resid bufC)
    run_layer(x,    IN, W0, al0, ar0, bufA, bufB, nullptr, 1);
    run_layer(bufB, HD, W1, al1, ar1, bufA, bufC, bufB,    1);
    run_layer(bufC, HD, W2, al2, ar2, bufA, bufB, bufC,    0);

    hipMemsetAsync(hg,  0, GG * DD * 4, stream);
    hipMemsetAsync(cnt, 0, GG * 4, stream);
    readout_kernel<<<N, 64, 0, stream>>>(bufB, gid, hg, cnt);
    classify_kernel<<<3, 256, 0, stream>>>(hg, cnt, Wc, bc, (float*)d_out);
}
